// Attention_9431748182753
// MI455X (gfx1250) — matrix operand order repaired and checked by reading, not compiled, not run on hardware
//
#include <hip/hip_runtime.h>
#include <hip/hip_bf16.h>
#include <math.h>

// Problem constants (from reference): B=32, N=1024, IN=64, ATTN=128, OUT=64, H=4
#define BATCH 32
#define NN    1024
#define CIN   64
#define CQK   128
#define COUT  64
#define CALL  320   // 128 Q + 128 K + 64 V

typedef __attribute__((ext_vector_type(2)))  float  v2f;
typedef __attribute__((ext_vector_type(8)))  float  v8f;
typedef __attribute__((ext_vector_type(16))) __bf16 v16bf;

union AF { v16bf v; unsigned int u[8]; };

// ---------------------------------------------------------------------------
// gfx1250 async global->LDS path (ASYNCcnt-tracked, no VGPR staging).
// Probe round 2 revealed the builtin's param type: int4-vector pointer in
// AS1 (global) / AS3 (LDS). Falls back to sync copies if builtins absent.
// ---------------------------------------------------------------------------
#if defined(__has_builtin)
#if __has_builtin(__builtin_amdgcn_global_load_async_to_lds_b128) && \
    __has_builtin(__builtin_amdgcn_s_wait_asynccnt)
#define HAVE_ASYNC_LDS 1
#endif
#endif
#ifndef HAVE_ASYNC_LDS
#define HAVE_ASYNC_LDS 0
#endif

typedef int v4i __attribute__((vector_size(16)));
typedef __attribute__((address_space(1))) v4i* gptr_v4i;
typedef __attribute__((address_space(3))) v4i* lptr_v4i;

__device__ __forceinline__ void cp16_async(void* lds, const void* gptr) {
#if HAVE_ASYNC_LDS
  __builtin_amdgcn_global_load_async_to_lds_b128((gptr_v4i)(gptr),
                                                 (lptr_v4i)(lds), 0, 0);
#else
  *(float4*)lds = *(const float4*)gptr;
#endif
}

__device__ __forceinline__ void async_wait_all() {
#if HAVE_ASYNC_LDS
  __builtin_amdgcn_s_wait_asynccnt(0);
#endif
}

__device__ __forceinline__ unsigned short f2bf(float f) {
  unsigned int x = __float_as_uint(f);
  x += 0x7fffu + ((x >> 16) & 1u);          // round-to-nearest-even
  return (unsigned short)(x >> 16);
}

// ---------------------------------------------------------------------------
// K1: degree normalization  dinv[b,n] = rsqrt(max(rowsum(adj with diag=1), 1))
// One wave32 per row; coalesced float4 reads; shuffle reduction.
// ---------------------------------------------------------------------------
__global__ void k1_degree(const float* __restrict__ adj, float* __restrict__ dinv) {
  const int row  = blockIdx.x * 8 + (threadIdx.x >> 5);   // global row in [0, B*N)
  const int lane = threadIdx.x & 31;
  const int b = row >> 10, n = row & (NN - 1);
  const float* p = adj + (size_t)b * NN * NN + (size_t)n * NN;
  float s = 0.f;
#pragma unroll
  for (int i = 0; i < 8; ++i) {
    __builtin_prefetch(p + (i * 32 + lane) * 4 + 512, 0, 1);  // global_prefetch_b8
    float4 v = *(const float4*)(p + (i * 32 + lane) * 4);
    s += v.x + v.y + v.z + v.w;
  }
#pragma unroll
  for (int off = 16; off > 0; off >>= 1) s += __shfl_xor(s, off, 32);
  if (lane == 0) {
    s = s - p[n] + 1.0f;                                  // self-loop: diag := 1
    dinv[row] = rsqrtf(fmaxf(s, 1.0f));
  }
}

// ---------------------------------------------------------------------------
// K2: H[m, 0:320] = dinv[m] * (x[m, :] @ [Wq | Wk | Wv])    (f32 WMMA 16x16x4)
// Block: 256 thr = 8 waves, tile 32 rows x 64 cols, K=64 staged in LDS.
// ---------------------------------------------------------------------------
__global__ void k2_features(const float* __restrict__ x,
                            const float* __restrict__ Wq,
                            const float* __restrict__ Wk,
                            const float* __restrict__ Wv,
                            const float* __restrict__ dinv,
                            float* __restrict__ H) {
  __shared__ float xs[32 * 65];
  __shared__ float ws[64 * 65];
  const int tid = threadIdx.x;
  const int m0 = blockIdx.x * 32;
  const int c0 = blockIdx.y * 64;
#pragma unroll
  for (int i = 0; i < 8; ++i) {                    // x tile 32x64
    int l = tid + i * 256, r = l >> 6, c = l & 63;
    xs[r * 65 + c] = x[(size_t)(m0 + r) * CIN + c];
  }
#pragma unroll
  for (int i = 0; i < 16; ++i) {                   // W tile 64x64 (concat select)
    int l = tid + i * 256, r = l >> 6, c = l & 63;
    int cg = c0 + c;
    float w;
    if (cg < 128)      w = Wq[r * 128 + cg];
    else if (cg < 256) w = Wk[r * 128 + (cg - 128)];
    else               w = Wv[r * 64 + (cg - 256)];
    ws[r * 65 + c] = w;
  }
  __syncthreads();

  const int lane = tid & 31, wid = tid >> 5;
  const int mt = wid >> 2, ct = wid & 3;           // 2 m-tiles x 4 c-tiles
  const int mrow = lane & 15, half = lane >> 4;
  v8f acc = {};
#pragma unroll
  for (int k0 = 0; k0 < 64; k0 += 4) {
    v2f a, bb;
    a.x = xs[(mt * 16 + mrow) * 65 + k0 + 2 * half];
    a.y = xs[(mt * 16 + mrow) * 65 + k0 + 2 * half + 1];
    bb.x = ws[(k0 + 2 * half) * 65 + ct * 16 + mrow];
    bb.y = ws[(k0 + 2 * half + 1) * 65 + ct * 16 + mrow];
    acc = __builtin_amdgcn_wmma_f32_16x16x4_f32(false, a, false, bb,
                                                (short)0, acc, false, false);
  }
#pragma unroll
  for (int r = 0; r < 8; ++r) {
    int m = m0 + mt * 16 + r + 8 * half;
    int c = c0 + ct * 16 + mrow;
    H[(size_t)m * CALL + c] = dinv[m] * acc[r];    // fold d[m] into H rows
  }
}

// ---------------------------------------------------------------------------
// K3: QKV[n,c] = dinv[n] * sum_m adjmod[n,m] * H[m,c] + bias[c]
// Per batch: M=1024, N=320, K=1024. Block tile 128x64, K-chunk 32.
// Double-buffered async global->LDS fills overlapped with f32 WMMA compute.
// Diagonal of adj patched in LDS after transfer. Q/K -> bf16 ws, V -> d_out.
// LDS: 2*128*36*4 + 2*32*68*4 = 54272 B.
// ---------------------------------------------------------------------------
__global__ void k3_gcn(const float* __restrict__ adj,
                       const float* __restrict__ H,
                       const float* __restrict__ dinv,
                       const float* __restrict__ bq,
                       const float* __restrict__ bk,
                       const float* __restrict__ bv,
                       unsigned short* __restrict__ qk,
                       float* __restrict__ vout) {
  __shared__ __align__(16) float As[2][128 * 36];   // stride 36: 144B rows (16B ok)
  __shared__ __align__(16) float Bs[2][32 * 68];    // stride 68: 272B rows (16B ok)
  const int tid = threadIdx.x;
  const int b  = blockIdx.z;
  const int n0 = blockIdx.x * 128;
  const int c0 = blockIdx.y * 64;
  const int lane = tid & 31, wid = tid >> 5;
  const int mrow = lane & 15, half = lane >> 4;
  const float* adjb = adj + (size_t)b * NN * NN;

  auto issue_chunk = [&](int ic) {
    const int mc = ic * 32;
    float* Ad = As[ic & 1];
    float* Bd = Bs[ic & 1];
#pragma unroll
    for (int i = 0; i < 4; ++i) {                  // A: 128x32 floats, B128 grains
      int l = tid + i * 256;
      int r = l >> 3, c4 = (l & 7) * 4;
      cp16_async(&Ad[r * 36 + c4], adjb + (size_t)(n0 + r) * NN + mc + c4);
    }
#pragma unroll
    for (int i = 0; i < 2; ++i) {                  // B: 32x64 floats from H
      int l = tid + i * 256;
      int r = l >> 4, c4 = (l & 15) * 4;
      cp16_async(&Bd[r * 68 + c4],
                 H + (size_t)(b * NN + mc + r) * CALL + c0 + c4);
    }
  };

  issue_chunk(0);
  v8f acc[4] = {};
  for (int ic = 0; ic < 32; ++ic) {
    const int mc = ic * 32;
    float* Ac = As[ic & 1];
    float* Bc = Bs[ic & 1];
    async_wait_all();                              // own transfers for chunk ic done
    __syncthreads();                               // everyone's transfers visible
    if (tid < 32) {                                // self-loop: diag := 1
      int n = mc + tid;
      if (n >= n0 && n < n0 + 128) Ac[(n - n0) * 36 + tid] = 1.0f;
    }
    __syncthreads();
    if (ic + 1 < 32) issue_chunk(ic + 1);          // overlap next fill with compute
#pragma unroll
    for (int kk = 0; kk < 32; kk += 4) {
      v2f a;
      a.x = Ac[(wid * 16 + mrow) * 36 + kk + 2 * half];
      a.y = Ac[(wid * 16 + mrow) * 36 + kk + 2 * half + 1];
#pragma unroll
      for (int ct = 0; ct < 4; ++ct) {
        v2f bb;
        bb.x = Bc[(kk + 2 * half) * 68 + ct * 16 + mrow];
        bb.y = Bc[(kk + 2 * half + 1) * 68 + ct * 16 + mrow];
        acc[ct] = __builtin_amdgcn_wmma_f32_16x16x4_f32(false, a, false, bb,
                                                        (short)0, acc[ct],
                                                        false, false);
      }
    }
  }
#pragma unroll
  for (int ct = 0; ct < 4; ++ct) {
#pragma unroll
    for (int r = 0; r < 8; ++r) {
      int n = n0 + wid * 16 + r + 8 * half;
      int c = c0 + ct * 16 + mrow;
      float bias = (c < 128) ? bq[c] : (c < 256) ? bk[c - 128] : bv[c - 256];
      float val = dinv[b * NN + n] * acc[ct][r] + bias;
      size_t row = (size_t)b * NN + n;
      if (c < 256) qk[row * 256 + c] = f2bf(val);          // Q|K as bf16
      else         vout[row * COUT + (c - 256)] = val;     // V output (fp32)
    }
  }
}

// ---------------------------------------------------------------------------
// K4: A = sym( mean_h tanh( Q_h K_h^T / 8 ) )   (bf16 WMMA 16x16x32)
// Block handles tile-pair (ib,jb), ib<=jb: computes T1=(i,j) and T2=(j,i),
// exchanges via LDS, writes both A halves coalesced. dh=32 == one WMMA K.
// Heads in 2 passes of 2; tile fills via async global->LDS (B128 grains).
// LDS: 4 tiles * 64 * 72 ushorts = 36864 B (T1/T2 overlaid after compute).
// ---------------------------------------------------------------------------
__global__ void k4_attention(const unsigned short* __restrict__ qk,
                             float* __restrict__ Aout) {
  __shared__ __align__(16) char smem[4 * 64 * 72 * 2];    // 36864 B
  unsigned short* tile0 = (unsigned short*)smem;
  // tiles: 0=Qi 1=Kj 2=Qj 3=Ki, each 64 rows x 64 cols (stride 72 ushorts=144B)
  float* T1 = (float*)smem;                               // overlaid after sync
  float* T2 = T1 + 64 * 65;                               // 2*64*65*4 = 33280 B

  const int tid = threadIdx.x;
  const int b = blockIdx.y;
  int t = blockIdx.x, ib = 0;
  while (t >= (16 - ib)) { t -= (16 - ib); ++ib; }
  const int jb = ib + t;
  const int i0 = ib * 64, j0 = jb * 64;

  const int lane = tid & 31, wid = tid >> 5;
  const int tsel = wid >> 2, mt = wid & 3;                // tsel: 0->T1, 1->T2
  const int mrow = lane & 15, half = lane >> 4;
  const unsigned short* qb = qk + (size_t)b * NN * 256;

  v8f tac[4] = {};
  for (int hp = 0; hp < 2; ++hp) {                        // head pairs
    if (hp) __syncthreads();                              // pass-0 reads done
#pragma unroll
    for (int i = 0; i < 8; ++i) {                         // 4 tiles, async B128
      int l = tid + i * 256;                              // 0..2047 grains
      int tt = l >> 9, w = l & 511;
      int r = w >> 3, c8 = (w & 7) * 8;                   // ushort col offset
      int rbase = (tt == 0 || tt == 3) ? i0 : j0;
      int cs = ((tt == 0 || tt == 2) ? 0 : 128) + hp * 64;
      cp16_async(tile0 + tt * 64 * 72 + r * 72 + c8,
                 qb + (size_t)(rbase + r) * 256 + cs + c8);
    }
    async_wait_all();
    __syncthreads();

    const unsigned int* Qu =
        (const unsigned int*)(tile0 + (tsel ? 2 : 0) * 64 * 72);
    const unsigned int* Ku =
        (const unsigned int*)(tile0 + (tsel ? 3 : 1) * 64 * 72);
    const int rowq = (mt * 16 + mrow) * 36;               // uint stride 36

    AF afr[2];
#pragma unroll
    for (int hh = 0; hh < 2; ++hh)
#pragma unroll
      for (int j = 0; j < 4; ++j) {
        afr[hh].u[j]     = Qu[rowq + hh * 16 + 4 * half + j];
        afr[hh].u[4 + j] = Qu[rowq + hh * 16 + 8 + 4 * half + j];
      }
#pragma unroll
    for (int nt = 0; nt < 4; ++nt) {
      const int colu = (nt * 16 + mrow) * 36;
#pragma unroll
      for (int hh = 0; hh < 2; ++hh) {
        AF bf;
#pragma unroll
        for (int v = 0; v < 8; ++v)
          bf.u[v] = Ku[colu + hh * 16 + (v >> 2) * 8 + 4 * half + (v & 3)];
        v8f s = {};
        s = __builtin_amdgcn_wmma_f32_16x16x32_bf16(false, afr[hh].v, false,
                                                    bf.v, (short)0, s,
                                                    false, false);
#pragma unroll
        for (int r = 0; r < 8; ++r)
          tac[nt][r] += 0.25f * tanhf(s[r] * 0.125f);     // mean over 4 heads
      }
    }
  }
  __syncthreads();                                        // done reading tiles
  {
    float* Tb = tsel ? T2 : T1;
#pragma unroll
    for (int nt = 0; nt < 4; ++nt)
#pragma unroll
      for (int r = 0; r < 8; ++r)
        Tb[(mt * 16 + r + 8 * half) * 65 + nt * 16 + mrow] = tac[nt][r];
  }
  __syncthreads();
  // A[i0+di, j0+dj] = A[j0+dj, i0+di] = 0.5*(T1[di,dj] + T2[dj,di])
#pragma unroll
  for (int i = 0; i < 16; ++i) {                          // upper tile, coalesced
    int l = tid + i * 256;
    int di = l >> 6, dj = l & 63;
    float v = 0.5f * (T1[di * 65 + dj] + T2[dj * 65 + di]);
    Aout[((size_t)b * NN + i0 + di) * NN + j0 + dj] = v;
  }
#pragma unroll
  for (int i = 0; i < 16; ++i) {                          // mirror tile, coalesced
    int l = tid + i * 256;
    int dj = l >> 6, di = l & 63;
    float v = 0.5f * (T1[di * 65 + dj] + T2[dj * 65 + di]);
    Aout[((size_t)b * NN + j0 + dj) * NN + i0 + di] = v;
  }
}

// ---------------------------------------------------------------------------
// Workspace layout (needs ~58.9 MB):
//   [0)            dinv   : 32768 f32            (128 KB)
//   [131072)       H      : 32768 x 320 f32      (41.94 MB)
//   [42074112)     qk     : 32768 x 256 bf16     (16.78 MB)
// ---------------------------------------------------------------------------
extern "C" void kernel_launch(void* const* d_in, const int* in_sizes, int n_in,
                              void* d_out, int out_size, void* d_ws, size_t ws_size,
                              hipStream_t stream) {
  (void)in_sizes; (void)n_in; (void)out_size; (void)ws_size;
  const float* x   = (const float*)d_in[0];
  const float* adj = (const float*)d_in[1];
  // d_in[2] = flags (unused by reference forward)
  const float* Wq = (const float*)d_in[3];
  const float* bq = (const float*)d_in[4];
  const float* Wk = (const float*)d_in[5];
  const float* bk = (const float*)d_in[6];
  const float* Wv = (const float*)d_in[7];
  const float* bv = (const float*)d_in[8];

  float* Vout = (float*)d_out;                              // [32,1024,64]
  float* Aout = (float*)d_out + (size_t)BATCH * NN * COUT;  // [32,1024,1024]

  char* ws = (char*)d_ws;
  float*          dinv = (float*)ws;
  float*          H    = (float*)(ws + 131072);
  unsigned short* qk   = (unsigned short*)(ws + 131072 + (size_t)32768 * 320 * 4);

  k1_degree  <<<dim3(BATCH * NN / 8), 256, 0, stream>>>(adj, dinv);
  k2_features<<<dim3(BATCH * NN / 32, CALL / 64), 256, 0, stream>>>(x, Wq, Wk, Wv, dinv, H);
  k3_gcn     <<<dim3(NN / 128, CALL / 64, BATCH), 256, 0, stream>>>(adj, H, dinv,
                                                                    bq, bk, bv, qk, Vout);
  k4_attention<<<dim3(136, BATCH), 256, 0, stream>>>(qk, Aout);
}
